// CrossAgentAttention_34282428956869
// MI455X (gfx1250) — matrix operand order repaired and checked by reading, not compiled, not run on hardware
//
#include <hip/hip_runtime.h>

// ---------------------------------------------------------------------------
// CrossAgentAttention fused kernel for MI455X (gfx1250, wave32, WMMA).
//
// Shapes: x (5,64,100,252) f32, Wqkv (768,64), bqkv (768), Wo (64,256), bo(64)
// out (5,64,100,252) f32.  HW = 25200 = 1575 tiles of 16 positions.
//
// Roofline: fused HBM traffic = read x (32MB) + write out (32MB) ~= 65 MB
// -> ~2.8us at 23.3 TB/s.  The 387MB qkv intermediate lives only in LDS
// (120KB f16 per 16-position tile).  GEMMs run on v_wmma_f32_16x16x32_f16.
// qkv LDS layout is [b][n][o] (channels contiguous) so the attention stages
// use b128 LDS accesses instead of scalar u16.
// ---------------------------------------------------------------------------

typedef __attribute__((ext_vector_type(16))) _Float16 v16h;
typedef __attribute__((ext_vector_type(8)))  _Float16 v8h;
typedef __attribute__((ext_vector_type(8)))  float    v8f;
typedef _Float16 half_t;

#define CAV       5
#define CCH       64          // C
#define NHEADS    4
#define INNER     256         // C * HEADS
#define QKV_ROWS  768         // 3 * INNER
#define HW        25200
#define TS        16          // spatial positions per workgroup
#define NTILES    (HW / TS)   // 1575

#define FRAG_H      512                     // halfs per 16x32 fragment (32 lanes * 16)
#define WQKV_FRAGS  (48 * 2)                // 768/16 mTiles * 64/32 kTiles
#define WO_FRAGS    (4 * 8)                 // 64/16 mTiles * 256/32 kTiles
#define PACK_WO_OFF (WQKV_FRAGS * FRAG_H)   // in halfs
#define PACK_TOTAL  ((WQKV_FRAGS + WO_FRAGS) * FRAG_H)   // 65536 halfs = 128 KB

// LDS carve (bytes)
#define OFF_BQKV   0                         // 768 f32
#define OFF_BO     3072                      // 64 f32
#define OFF_DIST   3328                      // 5 f32 (+pad)
#define OFF_BFRAGX 4096                      // 10 frags * 512 halfs = 10240 B
#define OFF_BFRAG2 14336                     // 40 frags * 512 halfs = 40960 B
#define OFF_QKV    55296                     // 5*16*768 halfs       = 122880 B
#define OFF_SCORE  178176                    // 16*4*5*5 f32         = 6400 B
#define SMEM_BYTES 184576

// A-fragment K mapping for 16-bit 16x32 A (contiguous, matches B fragments):
//   lane<16: j=0..15 -> K=j ;  lane>=16: K=j+16
__device__ __forceinline__ int kmapA(int lane, int j) {
    return j + 16 * (lane >> 4);
}

// ---------------------------------------------------------------------------
// Pre-pack Wqkv and Wo (f32 row-major) into f16 WMMA A-fragment order so the
// GEMM stages load each lane's fragment as contiguous 32B (2x global_load_b128).
// ---------------------------------------------------------------------------
__global__ void prepack_weights(const float* __restrict__ Wqkv,
                                const float* __restrict__ Wo,
                                half_t* __restrict__ packed) {
    int idx = blockIdx.x * 256 + threadIdx.x;          // 0 .. 65535
    if (idx < WQKV_FRAGS * FRAG_H) {
        int f = idx / FRAG_H, w = idx % FRAG_H;
        int lane = w >> 4, j = w & 15;
        int mT = f >> 1, kT = f & 1;
        int M = mT * 16 + (lane & 15);
        int K = kT * 32 + kmapA(lane, j);
        packed[idx] = (half_t)Wqkv[M * CCH + K];
    } else {
        int i2 = idx - WQKV_FRAGS * FRAG_H;
        int f = i2 / FRAG_H, w = i2 % FRAG_H;
        int lane = w >> 4, j = w & 15;
        int mT = f >> 3, kT = f & 7;
        int M = mT * 16 + (lane & 15);
        int K = kT * 32 + kmapA(lane, j);
        packed[idx] = (half_t)Wo[M * INNER + K];
    }
}

// ---------------------------------------------------------------------------
// Fused: qkv GEMM -> attention (score + context + softmax + AV) -> out GEMM.
// One workgroup (256 threads = 8 wave32) per 16 spatial positions.
// ---------------------------------------------------------------------------
__global__ __launch_bounds__(256)
void fused_cross_agent_attn(const float* __restrict__ x,
                            const float* __restrict__ dist,
                            const float* __restrict__ bqkv,
                            const float* __restrict__ bo,
                            const half_t* __restrict__ packed,
                            float* __restrict__ out) {
    extern __shared__ char smem[];
    float*  s_bqkv   = (float*)(smem + OFF_BQKV);
    float*  s_bo     = (float*)(smem + OFF_BO);
    float*  s_dist   = (float*)(smem + OFF_DIST);
    half_t* s_bfragx = (half_t*)(smem + OFF_BFRAGX);   // x tile as B fragments
    half_t* s_bfrag2 = (half_t*)(smem + OFF_BFRAG2);   // attn output as B fragments
    half_t* s_qkv    = (half_t*)(smem + OFF_QKV);      // qkv[b][n][o] f16 (o contiguous)
    float*  s_score  = (float*)(smem + OFF_SCORE);     // attn[s][h][b][d]

    const int tid  = threadIdx.x;
    const int lane = tid & 31;
    const int wave = tid >> 5;
    const int hi   = lane >> 4;      // which 16-lane half
    const int ln   = lane & 15;      // N within D tile
    const int s0   = blockIdx.x * TS;

    // L2-hot weight broadcast: warm this WGP's L0 with the packed fragments.
    __builtin_prefetch(packed + (size_t)tid * 128, 0, 0);                 // 64KB
    __builtin_prefetch(packed + 32768 + (size_t)tid * 128, 0, 0);         // 64KB

    // ---- stage 0: biases/distance to LDS; x tile -> B-fragment layout -----
    for (int i = tid; i < QKV_ROWS; i += 256) s_bqkv[i] = bqkv[i];
    if (tid < CCH) s_bo[tid]   = bo[tid];
    if (tid < CAV) s_dist[tid] = dist[tid];

    for (int rc = tid; rc < CAV * CCH; rc += 256) {      // one (b,c) row each
        int b = rc / CCH, c = rc % CCH;
        const float4* src = (const float4*)(x + (size_t)rc * HW + s0);
        float4 v0 = src[0], v1 = src[1], v2 = src[2], v3 = src[3];
        float vals[16] = {v0.x, v0.y, v0.z, v0.w, v1.x, v1.y, v1.z, v1.w,
                          v2.x, v2.y, v2.z, v2.w, v3.x, v3.y, v3.z, v3.w};
        int kT = c >> 5, K = c & 31;
        // B layout: pos = (N + 16*(K>=16))*16 + (K%16)
        half_t* dst = s_bfragx + ((b * 2 + kT) * FRAG_H) + 256 * (K >> 4) + (K & 15);
        #pragma unroll
        for (int N = 0; N < 16; ++N) dst[N * 16] = (half_t)vals[N];
    }
    __syncthreads();

    // ---- stage 1: qkv GEMM: D(16x16) = Wqkv_tile(16x32) x x_tile(32x16) ---
    // 240 tiles (b,mT), 2 chained WMMAs each, over 8 waves.
    for (int t = wave; t < CAV * 48; t += 8) {
        int b = t / 48, mT = t % 48;
        v8f acc = {};
        #pragma unroll
        for (int kT = 0; kT < 2; ++kT) {
            v16h a  = *(const v16h*)(packed + (mT * 2 + kT) * FRAG_H + lane * 16);
            v16h bb = *(const v16h*)(s_bfragx + (b * 2 + kT) * FRAG_H + lane * 16);
            acc = __builtin_amdgcn_wmma_f32_16x16x32_f16(
                false, a, false, bb, (short)0, acc, false, false);
        }
        // D: M = r + 8*hi (consecutive o), N = ln  ->  one b128 LDS store
        int o0 = mT * 16 + 8 * hi;
        v8h ov;
        #pragma unroll
        for (int r = 0; r < 8; ++r) ov[r] = (half_t)(acc[r] + s_bqkv[o0 + r]);
        *(v8h*)(s_qkv + (b * TS + ln) * QKV_ROWS + o0) = ov;
    }
    __syncthreads();

    // ---- stage 2: score + context + softmax (VALU; 5x5 per (s,h)) ---------
    // q/k rows are contiguous 64-half runs -> b128 LDS chunk loads.
    const float inv_sqrt_cav = 0.44721359549995793f;   // 1/sqrt(5)
    for (int it = tid; it < TS * NHEADS * CAV; it += 256) {
        int b = it % CAV;
        int h = (it / CAV) % NHEADS;
        int s = it / (CAV * NHEADS);
        const half_t* qbase = s_qkv + (b * TS + s) * QKV_ROWS + h * CCH;
        float acc[CAV] = {0.f, 0.f, 0.f, 0.f, 0.f};
        float qs = 0.f;
        #pragma unroll
        for (int ch = 0; ch < 8; ++ch) {               // 8 chunks of 8 channels
            v8h qv = *(const v8h*)(qbase + ch * 8);
            float qf[8];
            #pragma unroll
            for (int e = 0; e < 8; ++e) { qf[e] = (float)qv[e]; qs += qf[e]; }
            #pragma unroll
            for (int d = 0; d < CAV; ++d) {
                v8h kv = *(const v8h*)(s_qkv + (d * TS + s) * QKV_ROWS
                                       + INNER + h * CCH + ch * 8);
                #pragma unroll
                for (int e = 0; e < 8; ++e) acc[d] += qf[e] * (float)kv[e];
            }
        }
        float m = -1e30f;
        #pragma unroll
        for (int d = 0; d < CAV; ++d) {
            acc[d] = acc[d] * inv_sqrt_cav + qs * s_dist[d];
            m = fmaxf(m, acc[d]);
        }
        float sum = 0.f;
        #pragma unroll
        for (int d = 0; d < CAV; ++d) { acc[d] = __expf(acc[d] - m); sum += acc[d]; }
        float inv = 1.f / sum;
        float* sp = s_score + ((s * NHEADS + h) * CAV + b) * CAV;
        #pragma unroll
        for (int d = 0; d < CAV; ++d) sp[d] = acc[d] * inv;
    }
    __syncthreads();

    // ---- stage 3: out = attn @ v, deposited in B-fragment layout ----------
    // One thread per 8-channel chunk: 5x v8h loads + one v8h store.
    for (int it = tid; it < CAV * TS * (INNER / 8); it += 256) {
        int n  = it % TS;
        int oc = (it / TS) % (INNER / 8);
        int b  = it / (TS * (INNER / 8));
        int o0 = oc * 8;
        int h  = o0 >> 6;
        const float* ap = s_score + ((n * NHEADS + h) * CAV + b) * CAV;
        float acc8[8] = {0.f, 0.f, 0.f, 0.f, 0.f, 0.f, 0.f, 0.f};
        #pragma unroll
        for (int d = 0; d < CAV; ++d) {
            float a = ap[d];
            v8h vv = *(const v8h*)(s_qkv + (d * TS + n) * QKV_ROWS + 2 * INNER + o0);
            #pragma unroll
            for (int e = 0; e < 8; ++e) acc8[e] += a * (float)vv[e];
        }
        // B-frag: K = o%32; aligned 8-run keeps lane = n + 16*(K>>4) constant.
        int kT = o0 >> 5, K0 = o0 & 31;
        v8h ov;
        #pragma unroll
        for (int e = 0; e < 8; ++e) ov[e] = (half_t)acc8[e];
        *(v8h*)(s_bfrag2 + (b * 8 + kT) * FRAG_H
                + (n + 16 * (K0 >> 4)) * 16 + (K0 & 15)) = ov;
    }
    __syncthreads();

    // ---- stage 4: out GEMM: y(16x16) = Wo_tile(16x32) x attn_out(32x16) ---
    // 20 tiles (b,mT), 8 chained WMMAs each (K=256).
    for (int t = wave; t < CAV * 4; t += 8) {
        int b = t >> 2, mT = t & 3;
        v8f acc = {};
        #pragma unroll
        for (int kT = 0; kT < 8; ++kT) {
            v16h a  = *(const v16h*)(packed + PACK_WO_OFF + (mT * 8 + kT) * FRAG_H + lane * 16);
            v16h bb = *(const v16h*)(s_bfrag2 + (b * 8 + kT) * FRAG_H + lane * 16);
            acc = __builtin_amdgcn_wmma_f32_16x16x32_f16(
                false, a, false, bb, (short)0, acc, false, false);
        }
        #pragma unroll
        for (int r = 0; r < 8; ++r) {
            int o2 = mT * 16 + r + 8 * hi;
            out[(size_t)(b * CCH + o2) * HW + s0 + ln] = acc[r] + s_bo[o2];
        }
    }
}

extern "C" void kernel_launch(void* const* d_in, const int* in_sizes, int n_in,
                              void* d_out, int out_size, void* d_ws, size_t ws_size,
                              hipStream_t stream) {
    const float* x    = (const float*)d_in[0];
    const float* dist = (const float*)d_in[1];
    const float* Wqkv = (const float*)d_in[2];
    const float* bqkv = (const float*)d_in[3];
    const float* Wo   = (const float*)d_in[4];
    const float* bo   = (const float*)d_in[5];
    half_t* packed = (half_t*)d_ws;                      // 128 KB of workspace

    prepack_weights<<<PACK_TOTAL / 256, 256, 0, stream>>>(Wqkv, Wo, packed);
    fused_cross_agent_attn<<<NTILES, 256, SMEM_BYTES, stream>>>(
        x, dist, bqkv, bo, packed, (float*)d_out);
}
